// TransformerEncoder_44676249813605
// MI455X (gfx1250) — matrix operand order repaired and checked by reading, not compiled, not run on hardware
//
#include <hip/hip_runtime.h>
#include <math.h>

// ---------------- problem constants ----------------
#define BATCH    8
#define SEQ      1024
#define DMODEL   512
#define NHEADS   8
#define ATTN_U   64
#define FFU      2048
#define LN_EPS   1e-3f

// ---------------- types ----------------
typedef __bf16 bf16;
typedef __attribute__((ext_vector_type(16))) __bf16        v16bf;
typedef __attribute__((ext_vector_type(8)))  float         v8f;
typedef __attribute__((ext_vector_type(4)))  unsigned int  u32x4;   // 16B, trivial

#if defined(__gfx1250__) && __has_builtin(__builtin_amdgcn_tensor_load_to_lds)
#define USE_TDM 1
typedef __attribute__((ext_vector_type(8))) int i32x8;
typedef __attribute__((ext_vector_type(4))) int i32x4;
#else
#define USE_TDM 0
#endif

union ABfrag { v16bf v; u32x4 q[2]; };
union B8     { u32x4 v; bf16  e[8]; };

// ---------------- fp32 -> bf16 convert ----------------
__global__ __launch_bounds__(256)
void cvt_f32_bf16(const float* __restrict__ in, bf16* __restrict__ out, int n) {
  int i = blockIdx.x * blockDim.x + threadIdx.x;
  int stride = gridDim.x * blockDim.x;
  for (; i < n; i += stride) out[i] = (bf16)in[i];
}

// ---------------- tiled bf16 WMMA GEMM ----------------
// C[M,N] = A[M,K] * B[K,N]   (A bf16 row-major; B bf16 row-major, or if BT:
// logical B[k][n] = Bm[n*ldb + k]).
// Block tile 128x64, BK=32, 256 threads = 8 waves; wave (wm=wave>>1, wn=wave&1)
// owns a 32x32 output = 4 WMMA accumulators. M % 128 == 0, N % 64 == 0, K % 32 == 0.
template<bool BT, bool BIAS, bool RELU, bool OUTBF16>
__global__ __launch_bounds__(256)
void gemm_bf16(const bf16* __restrict__ A, const bf16* __restrict__ Bm,
               const float* __restrict__ bias,
               float* __restrict__ Cf, bf16* __restrict__ Cb,
               int M, int N, int K, int lda, int ldb, int ldc)
{
  constexpr int BM = 128, BN = 64, BK = 32, PAD = 8;   // PAD keeps 16B alignment
  __shared__ __align__(16) bf16 As[BM][BK + PAD];
  __shared__ __align__(16) bf16 Bs[BN][BK + PAD];      // stored [n][k] (K contiguous)

  const int tid  = threadIdx.x;
  const int lane = tid & 31;
  const int wave = tid >> 5;
  const int wm   = wave >> 1;         // 0..3 -> 32-row slice
  const int wn   = wave & 1;          // 0..1 -> 32-col slice
  const int l15  = lane & 15;
  const int hl   = lane >> 4;         // lane half (ISA fragment layouts)

  const int m0 = blockIdx.x * BM;
  const int n0 = blockIdx.y * BN;

  const v8f vzero = {0.f,0.f,0.f,0.f,0.f,0.f,0.f,0.f};
  v8f acc00 = vzero, acc01 = vzero, acc10 = vzero, acc11 = vzero;

  // ---- staging assignments (pointer-incremented, no per-iter 64-bit mul) ----
  const int ar = tid >> 2;            // A rows 0..63 (+64 for second load)
  const int ac = (tid & 3) * 8;       // A col group (8 bf16 = 16B)
  const bf16* aptr0 = A + (size_t)(m0 + ar) * lda + ac;
  const bf16* aptr1 = aptr0 + (size_t)64 * lda;
  (void)aptr0; (void)aptr1;

  const int bn_bt = tid >> 2;         // BT: n 0..63
  const int bk_bt = (tid & 3) * 8;    // BT: k group
  const int bk_n  = tid >> 3;         // !BT: k 0..31
  const int bn_n  = (tid & 7) * 8;    // !BT: n group
  const bf16* bptr = BT ? (Bm + (size_t)(n0 + bn_bt) * ldb + bk_bt)
                        : (Bm + (size_t)bk_n * ldb + n0 + bn_n);
  const size_t bstep = BT ? (size_t)BK : (size_t)BK * ldb;

#if USE_TDM
  // Tensor Data Mover: A tile is a 2D row-major tile (BK x BM in (x,y) = (K,M)),
  // DMA'd into LDS with padding matching As row stride. Wave 0 issues it.
  unsigned long long ga = (unsigned long long)(size_t)(A + (size_t)m0 * lda);
  const unsigned ldsA = (unsigned)(size_t)(&As[0][0]);
#endif

  for (int k0 = 0; k0 < K; k0 += BK) {
#if USE_TDM
    if (wave == 0) {
      u32x4 g0 = { 1u,                                   // count=1, user mode
                   ldsA,                                 // lds_addr
                   (unsigned)(ga & 0xffffffffull),       // global_addr[31:0]
                   (unsigned)((ga >> 32) & 0x01ffffffu) | (2u << 30) }; // [56:32], type=2
      i32x8 g1;
      g1[0] = (int)((1u << 16)      // data_size = 2 bytes
                  | (1u << 20)      // pad_enable
                  | (3u << 22)      // pad_interval: 16 DWORDs (= one 64B row)
                  | (3u << 25));    // pad_amount:   4 DWORDs (= 8 bf16 = PAD)
      g1[1] = (int)(((unsigned)K & 0xffffu) << 16);                       // tensor_dim0 lo
      g1[2] = (int)(((unsigned)K >> 16) | (((unsigned)M & 0xffffu) << 16)); // dim0 hi | dim1 lo
      g1[3] = (int)(((unsigned)M >> 16) | ((unsigned)BK << 16));          // dim1 hi | tile_dim0
      g1[4] = (int)(unsigned)BM;                                          // tile_dim1 (tile_dim2=0)
      g1[5] = (int)(unsigned)lda;                                         // tensor_dim0_stride lo
      g1[6] = 0;
      g1[7] = 0;
      i32x4 gz4 = {0,0,0,0};
      i32x8 gz8 = {0,0,0,0,0,0,0,0};
      __builtin_amdgcn_tensor_load_to_lds(g0, g1, gz4, gz4, gz8, 0);
      __builtin_amdgcn_s_wait_tensorcnt(0);
    }
    ga += (unsigned long long)BK * 2u;
#else
    *(u32x4*)&As[ar][ac]      = *(const u32x4*)aptr0;
    *(u32x4*)&As[ar + 64][ac] = *(const u32x4*)aptr1;
    aptr0 += BK; aptr1 += BK;
#endif

    if (BT) {
      *(u32x4*)&Bs[bn_bt][bk_bt] = *(const u32x4*)bptr;
    } else {
      B8 t; t.v = *(const u32x4*)bptr;     // coalesced 16B global load
      #pragma unroll
      for (int i = 0; i < 8; ++i) Bs[bn_n + i][bk_n] = t.e[i];  // LDS transpose
    }
    bptr += bstep;
    if (k0 + BK < K) __builtin_prefetch(bptr, 0, 1);
    __syncthreads();

    // ---- A fragments (16x32 bf16): lane M=l15, half hl.
    // VGPR j: K = (j>>2)*16 + hl*8 + (j&3)*2  -> two 16B LDS reads per fragment.
    ABfrag af0, af1, bf0, bf1;
    {
      const bf16* p = &As[wm * 32 + l15][hl * 8];
      af0.q[0] = *(const u32x4*)(p);
      af0.q[1] = *(const u32x4*)(p + 16);
      const bf16* q = &As[wm * 32 + 16 + l15][hl * 8];
      af1.q[0] = *(const u32x4*)(q);
      af1.q[1] = *(const u32x4*)(q + 16);
    }
    // ---- B fragments (32x16 bf16): lane N=l15, VGPR j: K = (j>>2)*16 + hl*8 + (j&3)*2.
    {
      const bf16* p = &Bs[wn * 32 + l15][hl * 8];
      bf0.q[0] = *(const u32x4*)(p);
      bf0.q[1] = *(const u32x4*)(p + 16);
      const bf16* q = &Bs[wn * 32 + 16 + l15][hl * 8];
      bf1.q[0] = *(const u32x4*)(q);
      bf1.q[1] = *(const u32x4*)(q + 16);
    }

    acc00 = __builtin_amdgcn_wmma_f32_16x16x32_bf16(false, af0.v, false, bf0.v,
                                                    (short)0, acc00, false, false);
    acc01 = __builtin_amdgcn_wmma_f32_16x16x32_bf16(false, af0.v, false, bf1.v,
                                                    (short)0, acc01, false, false);
    acc10 = __builtin_amdgcn_wmma_f32_16x16x32_bf16(false, af1.v, false, bf0.v,
                                                    (short)0, acc10, false, false);
    acc11 = __builtin_amdgcn_wmma_f32_16x16x32_bf16(false, af1.v, false, bf1.v,
                                                    (short)0, acc11, false, false);
    __syncthreads();
  }

  // ---- epilogue: C layout VGPR i, lane l: M = hl*8+i, N = l15 ----
  #pragma unroll
  for (int r = 0; r < 2; ++r) {
    #pragma unroll
    for (int c = 0; c < 2; ++c) {
      const v8f a = (r == 0) ? (c == 0 ? acc00 : acc01)
                             : (c == 0 ? acc10 : acc11);
      const int gn = n0 + wn * 32 + c * 16 + l15;
      const int gmb = m0 + wm * 32 + r * 16 + hl * 8;
      float bb = BIAS ? bias[gn] : 0.f;
      #pragma unroll
      for (int i = 0; i < 8; ++i) {
        float vv = a[i] + bb;
        if (RELU) vv = vv > 0.f ? vv : 0.f;
        if (OUTBF16) Cb[(size_t)(gmb + i) * ldc + gn] = (bf16)vv;
        else         Cf[(size_t)(gmb + i) * ldc + gn] = vv;
      }
    }
  }
}

// ---------------- fused scale + softmax (one row / block) ----------------
__global__ __launch_bounds__(256)
void softmax_row_bf16(const float* __restrict__ s, bf16* __restrict__ p, float scale)
{
  const int S = SEQ;
  const int row = blockIdx.x;
  const float* sr = s + (size_t)row * S;
  bf16* pr = p + (size_t)row * S;
  __shared__ float red[256];
  const int t = threadIdx.x;

  float x0 = sr[t]       * scale;
  float x1 = sr[t + 256] * scale;
  float x2 = sr[t + 512] * scale;
  float x3 = sr[t + 768] * scale;

  float m = fmaxf(fmaxf(x0, x1), fmaxf(x2, x3));
  red[t] = m; __syncthreads();
  for (int o = 128; o > 0; o >>= 1) { if (t < o) red[t] = fmaxf(red[t], red[t + o]); __syncthreads(); }
  m = red[0]; __syncthreads();

  float e0 = __expf(x0 - m), e1 = __expf(x1 - m), e2 = __expf(x2 - m), e3 = __expf(x3 - m);
  red[t] = e0 + e1 + e2 + e3; __syncthreads();
  for (int o = 128; o > 0; o >>= 1) { if (t < o) red[t] += red[t + o]; __syncthreads(); }
  const float inv = 1.f / red[0];

  pr[t]       = (bf16)(e0 * inv);
  pr[t + 256] = (bf16)(e1 * inv);
  pr[t + 512] = (bf16)(e2 * inv);
  pr[t + 768] = (bf16)(e3 * inv);
}

// ---------------- residual add + layernorm (one row / block, D=512) ----------------
__global__ __launch_bounds__(256)
void add_layernorm(const float* __restrict__ x, const float* __restrict__ y,
                   const float* __restrict__ gamma, const float* __restrict__ beta,
                   float* __restrict__ outf, bf16* __restrict__ outb)
{
  const int D = DMODEL;
  const int row = blockIdx.x;
  const int t = threadIdx.x;
  const float* xr = x + (size_t)row * D;
  const float* yr = y + (size_t)row * D;

  float v0 = xr[t] + yr[t];
  float v1 = xr[t + 256] + yr[t + 256];

  __shared__ float red[256];
  red[t] = v0 + v1; __syncthreads();
  for (int o = 128; o > 0; o >>= 1) { if (t < o) red[t] += red[t + o]; __syncthreads(); }
  const float mean = red[0] * (1.f / D); __syncthreads();

  const float d0 = v0 - mean, d1 = v1 - mean;
  red[t] = d0 * d0 + d1 * d1; __syncthreads();
  for (int o = 128; o > 0; o >>= 1) { if (t < o) red[t] += red[t + o]; __syncthreads(); }
  const float rstd = rsqrtf(red[0] * (1.f / D) + LN_EPS);

  const float o0 = gamma[t] * d0 * rstd + beta[t];
  const float o1 = gamma[t + 256] * d1 * rstd + beta[t + 256];
  outf[(size_t)row * D + t]       = o0;
  outf[(size_t)row * D + t + 256] = o1;
  if (outb) {
    outb[(size_t)row * D + t]       = (bf16)o0;
    outb[(size_t)row * D + t + 256] = (bf16)o1;
  }
}

// ---------------- host launcher ----------------
extern "C" void kernel_launch(void* const* d_in, const int* in_sizes, int n_in,
                              void* d_out, int out_size, void* d_ws, size_t ws_size,
                              hipStream_t stream) {
  (void)in_sizes; (void)n_in; (void)out_size; (void)ws_size;

  const float* x      = (const float*)d_in[0];
  const float* qw     = (const float*)d_in[1];
  const float* kw     = (const float*)d_in[2];
  const float* vw     = (const float*)d_in[3];
  const float* lw     = (const float*)d_in[4];
  const float* gamma1 = (const float*)d_in[5];
  const float* beta1  = (const float*)d_in[6];
  const float* w1     = (const float*)d_in[7];
  const float* b1     = (const float*)d_in[8];
  const float* w2     = (const float*)d_in[9];
  const float* b2     = (const float*)d_in[10];
  const float* gamma2 = (const float*)d_in[11];
  const float* beta2  = (const float*)d_in[12];

  char* ws = (char*)d_ws;
  size_t off = 0;
  auto alloc = [&](size_t bytes) -> char* {
    char* p = ws + off;
    off += (bytes + 255) & ~(size_t)255;
    return p;
  };

  const size_t BS = (size_t)BATCH * SEQ;
  bf16*  xb    = (bf16*) alloc(BS * DMODEL * 2);                    //   8.4 MB
  bf16*  qwb   = (bf16*) alloc((size_t)NHEADS * DMODEL * ATTN_U * 2);
  bf16*  kwb   = (bf16*) alloc((size_t)NHEADS * DMODEL * ATTN_U * 2);
  bf16*  vwb   = (bf16*) alloc((size_t)NHEADS * DMODEL * DMODEL * 2);
  bf16*  lwb   = (bf16*) alloc((size_t)NHEADS * DMODEL * DMODEL * 2);
  bf16*  w1b   = (bf16*) alloc((size_t)DMODEL * FFU * 2);
  bf16*  w2b   = (bf16*) alloc((size_t)FFU * DMODEL * 2);
  bf16*  qb    = (bf16*) alloc(BS * NHEADS * ATTN_U * 2);           //   8.4 MB
  bf16*  kb    = (bf16*) alloc(BS * NHEADS * ATTN_U * 2);
  bf16*  vb    = (bf16*) alloc(BS * NHEADS * DMODEL * 2);           //  67   MB
  float* sc    = (float*)alloc((size_t)SEQ * SEQ * 4);              //   4.2 MB (reused/head)
  bf16*  prb   = (bf16*) alloc((size_t)SEQ * SEQ * 2);              //   2.1 MB (reused/head)
  bf16*  attnb = (bf16*) alloc(BS * NHEADS * DMODEL * 2);           //  67   MB  [B,S,H*D]
  float* mha   = (float*)alloc(BS * DMODEL * 4);                    //  16.8 MB
  float* hf    = (float*)alloc(BS * DMODEL * 4);
  bf16*  hb    = (bf16*) alloc(BS * DMODEL * 2);
  bf16*  ff1b  = (bf16*) alloc(BS * FFU * 2);                       //  33.6 MB
  float* ff2   = (float*)alloc(BS * DMODEL * 4);
  // total ~272 MB

  // 1) downconvert activations + weights to bf16
  cvt_f32_bf16<<<2048, 256, 0, stream>>>(x,  xb,  (int)(BS * DMODEL));
  cvt_f32_bf16<<<256,  256, 0, stream>>>(qw, qwb, NHEADS * DMODEL * ATTN_U);
  cvt_f32_bf16<<<256,  256, 0, stream>>>(kw, kwb, NHEADS * DMODEL * ATTN_U);
  cvt_f32_bf16<<<1024, 256, 0, stream>>>(vw, vwb, NHEADS * DMODEL * DMODEL);
  cvt_f32_bf16<<<1024, 256, 0, stream>>>(lw, lwb, NHEADS * DMODEL * DMODEL);
  cvt_f32_bf16<<<512,  256, 0, stream>>>(w1, w1b, DMODEL * FFU);
  cvt_f32_bf16<<<512,  256, 0, stream>>>(w2, w2b, FFU * DMODEL);

  // 2) Q/K/V projections per (b,h)
  for (int b = 0; b < BATCH; ++b) {
    const bf16* Ax = xb + (size_t)b * SEQ * DMODEL;
    for (int h = 0; h < NHEADS; ++h) {
      const size_t bh = (size_t)b * NHEADS + h;
      gemm_bf16<false,false,false,true><<<dim3(SEQ/128, ATTN_U/64), 256, 0, stream>>>(
          Ax, qwb + (size_t)h * DMODEL * ATTN_U, nullptr,
          nullptr, qb + bh * SEQ * ATTN_U,
          SEQ, ATTN_U, DMODEL, DMODEL, ATTN_U, ATTN_U);
      gemm_bf16<false,false,false,true><<<dim3(SEQ/128, ATTN_U/64), 256, 0, stream>>>(
          Ax, kwb + (size_t)h * DMODEL * ATTN_U, nullptr,
          nullptr, kb + bh * SEQ * ATTN_U,
          SEQ, ATTN_U, DMODEL, DMODEL, ATTN_U, ATTN_U);
      gemm_bf16<false,false,false,true><<<dim3(SEQ/128, DMODEL/64), 256, 0, stream>>>(
          Ax, vwb + (size_t)h * DMODEL * DMODEL, nullptr,
          nullptr, vb + bh * SEQ * DMODEL,
          SEQ, DMODEL, DMODEL, DMODEL, DMODEL, DMODEL);
    }
  }

  // 3) attention per (b,h): scores -> softmax -> probs @ V (into concat layout)
  const float scale = 1.0f / sqrtf((float)DMODEL);
  for (int b = 0; b < BATCH; ++b) {
    for (int h = 0; h < NHEADS; ++h) {
      const size_t bh = (size_t)b * NHEADS + h;
      // scores = Q (S x U) * K^T (U x S), fp32
      gemm_bf16<true,false,false,false><<<dim3(SEQ/128, SEQ/64), 256, 0, stream>>>(
          qb + bh * SEQ * ATTN_U, kb + bh * SEQ * ATTN_U, nullptr,
          sc, nullptr,
          SEQ, SEQ, ATTN_U, ATTN_U, ATTN_U, SEQ);
      softmax_row_bf16<<<SEQ, 256, 0, stream>>>(sc, prb, scale);
      // attn = probs (S x S) * V (S x D) -> attnb[b, :, h*D : h*D+D]
      gemm_bf16<false,false,false,true><<<dim3(SEQ/128, DMODEL/64), 256, 0, stream>>>(
          prb, vb + bh * SEQ * DMODEL, nullptr,
          nullptr, attnb + (size_t)b * SEQ * (NHEADS * DMODEL) + (size_t)h * DMODEL,
          SEQ, DMODEL, SEQ, SEQ, DMODEL, NHEADS * DMODEL);
    }
  }

  // 4) output projection: [B*S, H*D] @ [H*D, D] -> mha
  gemm_bf16<false,false,false,false><<<dim3((unsigned)(BS/128), DMODEL/64), 256, 0, stream>>>(
      attnb, lwb, nullptr, mha, nullptr,
      (int)BS, DMODEL, NHEADS * DMODEL, NHEADS * DMODEL, DMODEL, DMODEL);

  // 5) h = LN1(x + mha)   (fp32 + bf16 copies)
  add_layernorm<<<(unsigned)BS, 256, 0, stream>>>(x, mha, gamma1, beta1, hf, hb);

  // 6) ff1 = relu(h @ w1 + b1)  (bf16 out)
  gemm_bf16<false,true,true,true><<<dim3((unsigned)(BS/128), FFU/64), 256, 0, stream>>>(
      hb, w1b, b1, nullptr, ff1b,
      (int)BS, FFU, DMODEL, DMODEL, FFU, FFU);

  // 7) ff2 = ff1 @ w2 + b2  (fp32 out)
  gemm_bf16<false,true,false,false><<<dim3((unsigned)(BS/128), DMODEL/64), 256, 0, stream>>>(
      ff1b, w2b, b2, ff2, nullptr,
      (int)BS, DMODEL, FFU, FFU, DMODEL, DMODEL);

  // 8) out = LN2(h + ff2)
  add_layernorm<<<(unsigned)BS, 256, 0, stream>>>(hf, ff2, gamma2, beta2,
                                                  (float*)d_out, nullptr);
}
